// GNNModel_59596966199718
// MI455X (gfx1250) — compile-verified, matrix operand order checked by reading, not run on hardware
//
#include <hip/hip_runtime.h>
#include <hip/hip_bf16.h>

typedef __attribute__((ext_vector_type(16))) _Float16 v16h;
typedef __attribute__((ext_vector_type(8)))  _Float16 v8h;
typedef __attribute__((ext_vector_type(8)))  float    v8f;

#define HDIM  128
#define INC   32
#define NIP   20000
#define NPP   50000
#define NFL   100000
#define NEDGE 200000
#define NITER 4

#define LDM   264   // padded LDS row (halves) for 256-wide W_msg rows: 528B -> 4-bank lane stride
#define LDG   136   // padded LDS row (halves) for 128-wide GRU weight rows: 272B -> 4-bank stride

// ---------------------------------------------------------------- helpers

__device__ __forceinline__ v8f splat8(float v) {
  v8f r;
#pragma unroll
  for (int i = 0; i < 8; i++) r[i] = v;
  return r;
}

// CDNA5 16-bit A/B fragment from one row of a row-major f16 matrix.
// halves[0..7]  = row[k0 + hi*8 .. +7]
// halves[8..15] = row[k0 + 16 + hi*8 .. +7]
__device__ __forceinline__ v16h frag_from_row(const _Float16* row, int k0, int hi) {
  v8h lo = *(const v8h*)(row + k0 + hi * 8);
  v8h hb = *(const v8h*)(row + k0 + 16 + hi * 8);
  v16h r;
#pragma unroll
  for (int i = 0; i < 8; i++) { r[i] = lo[i]; r[8 + i] = hb[i]; }
  return r;
}

// A fragment: rows r0..r0+15 of row-major [*,ld]; lane&15 selects row.
// B fragment: same loader on W stored [N][K] row-major (K contiguous).
__device__ __forceinline__ v16h frag_rm(const _Float16* M, int ld, int r0, int k0, int lane) {
  return frag_from_row(M + (size_t)(r0 + (lane & 15)) * ld, k0, lane >> 4);
}

__device__ __forceinline__ v8f wmma16(v16h a, v16h b, v8f c) {
  return __builtin_amdgcn_wmma_f32_16x16x32_f16(false, a, false, b, (short)0, c, false, false);
}

__device__ __forceinline__ float sigm(float x) { return 1.0f / (1.0f + __expf(-x)); }

// Async global->LDS copy of 16 bytes per lane (ASYNCcnt path).
__device__ __forceinline__ void async_copy_b128(unsigned lds_off, const void* gptr) {
  asm volatile("global_load_async_to_lds_b128 %0, %1, off"
               : : "v"(lds_off), "v"((unsigned long long)(uintptr_t)gptr) : "memory");
}
__device__ __forceinline__ void async_wait0() {
  asm volatile("s_wait_asynccnt 0x0" : : : "memory");
}

// ---------------------------------------------------------------- elementwise

__global__ void k_f32_to_f16(const float* __restrict__ x, _Float16* __restrict__ y, int n) {
  int i = blockIdx.x * blockDim.x + threadIdx.x;
  if (i < n) y[i] = (_Float16)x[i];
}

__global__ void k_copy_init(const float* __restrict__ x, float* __restrict__ yf,
                            _Float16* __restrict__ yh, int n) {
  int i = blockIdx.x * blockDim.x + threadIdx.x;
  if (i < n) { float v = x[i]; yf[i] = v; yh[i] = (_Float16)v; }
}

__global__ void k_pad_flow(const float* __restrict__ x, float* __restrict__ yf,
                           _Float16* __restrict__ yh) {
  int i = blockIdx.x * blockDim.x + threadIdx.x;
  if (i < NFL * HDIM) {
    int c = i & (HDIM - 1);
    int r = i >> 7;
    float v = (c < INC) ? x[r * INC + c] : 0.0f;
    yf[i] = v; yh[i] = (_Float16)v;
  }
}

__global__ void k_count(const int* __restrict__ dst, float* __restrict__ cnt, int ne) {
  int i = blockIdx.x * blockDim.x + threadIdx.x;
  if (i < ne) atomicAdd(&cnt[dst[i]], 1.0f);
}

__global__ void k_norm(const float* __restrict__ acc, const float* __restrict__ cnt,
                       _Float16* __restrict__ m, int total) {
  int i = blockIdx.x * blockDim.x + threadIdx.x;
  if (i < total) {
    float c = fmaxf(cnt[i >> 7], 1.0f);
    m[i] = (_Float16)(acc[i] / c);
  }
}

// ---------------------------------------------------------------- messages
// msg[e,:] = concat(h_src[src[e]], h_dst[dst[e]]) @ Wm^T + bm   (Wm: [H, 2H])
// scatter-add into acc[dst[e], :]. One wave = one 16-edge tile, grid-stride.
// Wm staged once per block in LDS (rows padded to 264 halves: conflict-free).

__global__ void __launch_bounds__(256)
k_msg(const _Float16* __restrict__ hsrc, const _Float16* __restrict__ hdst,
      const int* __restrict__ src, const int* __restrict__ dst,
      const _Float16* __restrict__ Wm, const float* __restrict__ bm,
      float* __restrict__ acc, int ntiles) {
  __shared__ __align__(16) _Float16 sW[HDIM * LDM];   // 128 rows x 264 halves = 66KB

  {
    unsigned base = (unsigned)(uintptr_t)&sW[0];
    // 128 rows x 32 chunks of 8 halves
#pragma unroll 4
    for (int c = threadIdx.x; c < HDIM * 32; c += 256) {
      int row = c >> 5, off = c & 31;
      async_copy_b128(base + (unsigned)(row * LDM + off * 8) * 2u,
                      Wm + (size_t)row * (2 * HDIM) + off * 8);
    }
    async_wait0();
  }
  __syncthreads();

  int wave = threadIdx.x >> 5, lane = threadIdx.x & 31;
  int m = lane & 15, hi = lane >> 4;

  for (int tile = blockIdx.x * 8 + wave; tile < ntiles; tile += gridDim.x * 8) {
    int e0 = tile * 16;
    const _Float16* rs = hsrc + (size_t)src[e0 + m] * HDIM;
    const _Float16* rd = hdst + (size_t)dst[e0 + m] * HDIM;

    v16h a[8];
#pragma unroll
    for (int kk = 0; kk < 4; kk++) a[kk] = frag_from_row(rs, kk * 32, hi);
#pragma unroll
    for (int kk = 0; kk < 4; kk++) a[4 + kk] = frag_from_row(rd, kk * 32, hi);

    int didx[8];
#pragma unroll
    for (int r = 0; r < 8; r++) didx[r] = dst[e0 + hi * 8 + r];

    for (int nt = 0; nt < 8; nt++) {
      int n = nt * 16 + m;
      v8f c = splat8(bm[n]);
#pragma unroll
      for (int kk = 0; kk < 8; kk++)
        c = wmma16(a[kk], frag_rm(sW, LDM, nt * 16, kk * 32, lane), c);
#pragma unroll
      for (int r = 0; r < 8; r++)
        atomicAdd(&acc[(size_t)didx[r] * HDIM + n], c[r]);
    }
  }
}

// ---------------------------------------------------------------- GRU
// h' = (1-z)*tanh(in + r*hn) + z*h, gates from x@wih^T and h@whh^T.
// 512-thread blocks: both weight matrices (384x128 each) staged padded in LDS
// (2 x 384 x 136 halves = 204KB, within the 320KB WGP LDS), then 16 waves
// grid-stride over 16-node tiles. Kills the dominant L2 weight re-read
// (192KB per tile otherwise).

__global__ void __launch_bounds__(512)
k_gru(const _Float16* __restrict__ x16, float* __restrict__ hf, _Float16* __restrict__ hh,
      const _Float16* __restrict__ wih, const _Float16* __restrict__ whh,
      const float* __restrict__ bih, const float* __restrict__ bhh, int ntiles) {
  __shared__ __align__(16) _Float16 sw[2 * 3 * HDIM * LDG];  // wih then whh, padded rows

  {
    unsigned base = (unsigned)(uintptr_t)&sw[0];
    // 2 matrices x 384 rows x 16 chunks of 8 halves = 12288 chunks
#pragma unroll 4
    for (int c = threadIdx.x; c < 2 * 3 * HDIM * 16; c += 512) {
      int mat = c >> 13;             // 0: wih, 1: whh   (6144 chunks each)
      int rc = c & 8191;
      int row = rc >> 4, off = rc & 15;
      const _Float16* g = (mat ? whh : wih) + (size_t)row * HDIM + off * 8;
      async_copy_b128(base + (unsigned)(mat * 3 * HDIM * LDG + row * LDG + off * 8) * 2u, g);
    }
    async_wait0();
  }
  __syncthreads();

  const _Float16* sih = sw;
  const _Float16* shh = sw + 3 * HDIM * LDG;

  int wave = threadIdx.x >> 5, lane = threadIdx.x & 31;
  int m = lane & 15, hi = lane >> 4;

  for (int tile = blockIdx.x * 16 + wave; tile < ntiles; tile += gridDim.x * 16) {
    int r0 = tile * 16;

    v16h ax[4], ah[4];
#pragma unroll
    for (int kk = 0; kk < 4; kk++) {
      ax[kk] = frag_rm(x16, HDIM, r0, kk * 32, lane);
      ah[kk] = frag_rm(hh, HDIM, r0, kk * 32, lane);
    }

    for (int nt = 0; nt < 8; nt++) {
      int n = nt * 16 + m;
      v8f cr = splat8(bih[n] + bhh[n]);                 // r-gate (both matmuls)
      v8f cz = splat8(bih[HDIM + n] + bhh[HDIM + n]);   // z-gate (both matmuls)
      v8f ci = splat8(bih[2 * HDIM + n]);               // i-part of n-gate
      v8f ch = splat8(bhh[2 * HDIM + n]);               // h-part of n-gate
#pragma unroll
      for (int kk = 0; kk < 4; kk++) {
        v16h b0 = frag_rm(sih, LDG, nt * 16, kk * 32, lane);
        v16h b1 = frag_rm(shh, LDG, nt * 16, kk * 32, lane);
        v16h b2 = frag_rm(sih, LDG, HDIM + nt * 16, kk * 32, lane);
        v16h b3 = frag_rm(shh, LDG, HDIM + nt * 16, kk * 32, lane);
        v16h b4 = frag_rm(sih, LDG, 2 * HDIM + nt * 16, kk * 32, lane);
        v16h b5 = frag_rm(shh, LDG, 2 * HDIM + nt * 16, kk * 32, lane);
        cr = wmma16(ax[kk], b0, cr);
        cr = wmma16(ah[kk], b1, cr);
        cz = wmma16(ax[kk], b2, cz);
        cz = wmma16(ah[kk], b3, cz);
        ci = wmma16(ax[kk], b4, ci);
        ch = wmma16(ah[kk], b5, ch);
      }
#pragma unroll
      for (int r = 0; r < 8; r++) {
        size_t idx = (size_t)(r0 + hi * 8 + r) * HDIM + n;
        float hold = hf[idx];
        float rg = sigm(cr[r]);
        float zg = sigm(cz[r]);
        float ng = tanhf(ci[r] + rg * ch[r]);
        float hnew = (1.0f - zg) * ng + zg * hold;
        hf[idx] = hnew;
        hh[idx] = (_Float16)hnew;
      }
    }
  }
}

// ---------------------------------------------------------------- readout
// relu(h@W1^T+b1) -> relu(@W2^T+b2) -> @W3^T+b3 -> softmax(2).
// One wave = 16 flow nodes; LDS round-trip (padded rows, same-wave DS order).

#define LD1 136   // padded s1 row (halves)
#define LD2 72    // padded s2 row (halves)

__global__ void __launch_bounds__(128)
k_readout(const _Float16* __restrict__ hfl, const _Float16* __restrict__ W1h,
          const float* __restrict__ b1, const _Float16* __restrict__ W2h,
          const float* __restrict__ b2, const float* __restrict__ W3,
          const float* __restrict__ b3, float* __restrict__ out, int ntiles) {
  __shared__ __align__(16) _Float16 s1[4][16 * LD1];
  __shared__ __align__(16) _Float16 s2[4][16 * LD2];
  int wave = threadIdx.x >> 5, lane = threadIdx.x & 31;
  int tile = blockIdx.x * 4 + wave;
  if (tile >= ntiles) return;
  int r0 = tile * 16;
  int m = lane & 15, hi = lane >> 4;

  v16h a[4];
#pragma unroll
  for (int kk = 0; kk < 4; kk++) a[kk] = frag_rm(hfl, HDIM, r0, kk * 32, lane);

  for (int nt = 0; nt < 8; nt++) {
    v8f c = splat8(b1[nt * 16 + m]);
#pragma unroll
    for (int kk = 0; kk < 4; kk++)
      c = wmma16(a[kk], frag_rm(W1h, HDIM, nt * 16, kk * 32, lane), c);
#pragma unroll
    for (int r = 0; r < 8; r++)
      s1[wave][(hi * 8 + r) * LD1 + nt * 16 + m] = (_Float16)fmaxf(c[r], 0.0f);
  }

  v16h a2[4];
#pragma unroll
  for (int kk = 0; kk < 4; kk++) a2[kk] = frag_rm(&s1[wave][0], LD1, 0, kk * 32, lane);

  for (int nt = 0; nt < 4; nt++) {
    v8f c = splat8(b2[nt * 16 + m]);
#pragma unroll
    for (int kk = 0; kk < 4; kk++)
      c = wmma16(a2[kk], frag_rm(W2h, HDIM, nt * 16, kk * 32, lane), c);
#pragma unroll
    for (int r = 0; r < 8; r++)
      s2[wave][(hi * 8 + r) * LD2 + nt * 16 + m] = (_Float16)fmaxf(c[r], 0.0f);
  }

  if (lane < 16) {
    float l0 = b3[0], l1 = b3[1];
    for (int c = 0; c < 64; c++) {
      float v = (float)s2[wave][lane * LD2 + c];
      l0 += v * W3[c];
      l1 += v * W3[64 + c];
    }
    float mx = fmaxf(l0, l1);
    float e0 = __expf(l0 - mx), e1 = __expf(l1 - mx);
    float s = e0 + e1;
    out[(size_t)(r0 + lane) * 2 + 0] = e0 / s;
    out[(size_t)(r0 + lane) * 2 + 1] = e1 / s;
  }
}

// ---------------------------------------------------------------- launch

static inline size_t aln(size_t x) { return (x + 255) & ~(size_t)255; }
static inline unsigned umin(unsigned a, unsigned b) { return a < b ? a : b; }

extern "C" void kernel_launch(void* const* d_in, const int* in_sizes, int n_in,
                              void* d_out, int out_size, void* d_ws, size_t ws_size,
                              hipStream_t stream) {
  const float* x_ip   = (const float*)d_in[0];
  const float* x_pp   = (const float*)d_in[1];
  const float* x_fl   = (const float*)d_in[2];
  const float* Wmsg_ip = (const float*)d_in[3];
  const float* bmsg_ip = (const float*)d_in[4];
  const float* Wmsg_fl = (const float*)d_in[5];
  const float* bmsg_fl = (const float*)d_in[6];
  const float* wih_ip = (const float*)d_in[7];
  const float* whh_ip = (const float*)d_in[8];
  const float* bih_ip = (const float*)d_in[9];
  const float* bhh_ip = (const float*)d_in[10];
  const float* wih_fl = (const float*)d_in[11];
  const float* whh_fl = (const float*)d_in[12];
  const float* bih_fl = (const float*)d_in[13];
  const float* bhh_fl = (const float*)d_in[14];
  const float* W1 = (const float*)d_in[15];
  const float* b1 = (const float*)d_in[16];
  const float* W2 = (const float*)d_in[17];
  const float* b2 = (const float*)d_in[18];
  const float* W3 = (const float*)d_in[19];
  const float* b3 = (const float*)d_in[20];
  const int* src_ip2pp = (const int*)d_in[21];
  const int* dst_ip2pp = (const int*)d_in[22];
  const int* src_pp2ip = (const int*)d_in[23];
  const int* dst_pp2ip = (const int*)d_in[24];
  const int* src_pp2fl = (const int*)d_in[25];
  const int* dst_pp2fl = (const int*)d_in[26];
  const int* src_fl2pp = (const int*)d_in[27];
  const int* dst_fl2pp = (const int*)d_in[28];
  float* out = (float*)d_out;

  // ---- workspace carve-up
  char* w = (char*)d_ws;
  auto alloc = [&](size_t bytes) -> char* { char* p = w; w += aln(bytes); return p; };

  float*    h_ip  = (float*)   alloc((size_t)NIP * HDIM * 4);
  float*    h_pp  = (float*)   alloc((size_t)NPP * HDIM * 4);
  float*    h_fl  = (float*)   alloc((size_t)NFL * HDIM * 4);
  _Float16* h_ip16 = (_Float16*)alloc((size_t)NIP * HDIM * 2);
  _Float16* h_pp16 = (_Float16*)alloc((size_t)NPP * HDIM * 2);
  _Float16* h_fl16 = (_Float16*)alloc((size_t)NFL * HDIM * 2);

  size_t acc_floats = (size_t)(NPP + NIP + NFL + NPP) * HDIM;
  float* acc_all = (float*)alloc(acc_floats * 4);
  float* acc_ip2pp = acc_all;                          // [NPP,H]
  float* acc_pp2ip = acc_ip2pp + (size_t)NPP * HDIM;   // [NIP,H]
  float* acc_pp2fl = acc_pp2ip + (size_t)NIP * HDIM;   // [NFL,H]
  float* acc_fl2pp = acc_pp2fl + (size_t)NFL * HDIM;   // [NPP,H]

  _Float16* m_all = (_Float16*)alloc(acc_floats * 2);
  _Float16* m_ip2pp = m_all;
  _Float16* m_pp2ip = m_ip2pp + (size_t)NPP * HDIM;
  _Float16* m_pp2fl = m_pp2ip + (size_t)NIP * HDIM;
  _Float16* m_fl2pp = m_pp2fl + (size_t)NFL * HDIM;

  size_t cnt_floats = (size_t)(NPP + NIP + NFL + NPP);
  float* cnt_all = (float*)alloc(cnt_floats * 4);
  float* cnt_ip2pp = cnt_all;
  float* cnt_pp2ip = cnt_ip2pp + NPP;
  float* cnt_pp2fl = cnt_pp2ip + NIP;
  float* cnt_fl2pp = cnt_pp2fl + NFL;

  _Float16* Wmsg_ip16 = (_Float16*)alloc(HDIM * 2 * HDIM * 2);
  _Float16* Wmsg_fl16 = (_Float16*)alloc(HDIM * 2 * HDIM * 2);
  _Float16* wih_ip16  = (_Float16*)alloc(3 * HDIM * HDIM * 2);
  _Float16* whh_ip16  = (_Float16*)alloc(3 * HDIM * HDIM * 2);
  _Float16* wih_fl16  = (_Float16*)alloc(3 * HDIM * HDIM * 2);
  _Float16* whh_fl16  = (_Float16*)alloc(3 * HDIM * HDIM * 2);
  _Float16* W1h       = (_Float16*)alloc(HDIM * HDIM * 2);
  _Float16* W2h       = (_Float16*)alloc(64 * HDIM * 2);

  auto g1 = [](size_t n) { return dim3((unsigned)((n + 255) / 256)); };

  // ---- one-time per call: counts (edge lists are iteration-invariant)
  hipMemsetAsync(cnt_all, 0, cnt_floats * 4, stream);
  k_count<<<g1(NEDGE), 256, 0, stream>>>(dst_ip2pp, cnt_ip2pp, NEDGE);
  k_count<<<g1(NEDGE), 256, 0, stream>>>(dst_pp2ip, cnt_pp2ip, NEDGE);
  k_count<<<g1(NEDGE), 256, 0, stream>>>(dst_pp2fl, cnt_pp2fl, NEDGE);
  k_count<<<g1(NEDGE), 256, 0, stream>>>(dst_fl2pp, cnt_fl2pp, NEDGE);

  // ---- state init
  k_copy_init<<<g1((size_t)NIP * HDIM), 256, 0, stream>>>(x_ip, h_ip, h_ip16, NIP * HDIM);
  k_copy_init<<<g1((size_t)NPP * HDIM), 256, 0, stream>>>(x_pp, h_pp, h_pp16, NPP * HDIM);
  k_pad_flow<<<g1((size_t)NFL * HDIM), 256, 0, stream>>>(x_fl, h_fl, h_fl16);

  // ---- weight f16 copies
  k_f32_to_f16<<<g1(HDIM * 2 * HDIM), 256, 0, stream>>>(Wmsg_ip, Wmsg_ip16, HDIM * 2 * HDIM);
  k_f32_to_f16<<<g1(HDIM * 2 * HDIM), 256, 0, stream>>>(Wmsg_fl, Wmsg_fl16, HDIM * 2 * HDIM);
  k_f32_to_f16<<<g1(3 * HDIM * HDIM), 256, 0, stream>>>(wih_ip, wih_ip16, 3 * HDIM * HDIM);
  k_f32_to_f16<<<g1(3 * HDIM * HDIM), 256, 0, stream>>>(whh_ip, whh_ip16, 3 * HDIM * HDIM);
  k_f32_to_f16<<<g1(3 * HDIM * HDIM), 256, 0, stream>>>(wih_fl, wih_fl16, 3 * HDIM * HDIM);
  k_f32_to_f16<<<g1(3 * HDIM * HDIM), 256, 0, stream>>>(whh_fl, whh_fl16, 3 * HDIM * HDIM);
  k_f32_to_f16<<<g1(HDIM * HDIM), 256, 0, stream>>>(W1, W1h, HDIM * HDIM);
  k_f32_to_f16<<<g1(64 * HDIM), 256, 0, stream>>>(W2, W2h, 64 * HDIM);

  const int etiles = NEDGE / 16;                 // 12500
  const int tip = NIP / 16, tpp = NPP / 16, tfl = NFL / 16;

  const dim3 eblk(umin((etiles + 7) / 8, 768)), msg_tpb(256);
  const dim3 gip(umin((tip + 15) / 16, 160)), gpp(umin((tpp + 15) / 16, 160)),
             gfl(umin((tfl + 15) / 16, 160)), gru_tpb(512);

  // n_iter is fixed to 4 by the reference setup (cannot sync-read a device
  // scalar during graph capture).
  for (int it = 0; it < NITER; it++) {
    hipMemsetAsync(acc_all, 0, acc_floats * 4, stream);

    // all 4 messages from pre-update state
    k_msg<<<eblk, msg_tpb, 0, stream>>>(h_ip16, h_pp16, src_ip2pp, dst_ip2pp,
                                        Wmsg_ip16, bmsg_ip, acc_ip2pp, etiles);
    k_msg<<<eblk, msg_tpb, 0, stream>>>(h_pp16, h_ip16, src_pp2ip, dst_pp2ip,
                                        Wmsg_ip16, bmsg_ip, acc_pp2ip, etiles);
    k_msg<<<eblk, msg_tpb, 0, stream>>>(h_pp16, h_fl16, src_pp2fl, dst_pp2fl,
                                        Wmsg_fl16, bmsg_fl, acc_pp2fl, etiles);
    k_msg<<<eblk, msg_tpb, 0, stream>>>(h_fl16, h_pp16, src_fl2pp, dst_fl2pp,
                                        Wmsg_fl16, bmsg_fl, acc_fl2pp, etiles);

    k_norm<<<g1((size_t)NPP * HDIM), 256, 0, stream>>>(acc_ip2pp, cnt_ip2pp, m_ip2pp, NPP * HDIM);
    k_norm<<<g1((size_t)NIP * HDIM), 256, 0, stream>>>(acc_pp2ip, cnt_pp2ip, m_pp2ip, NIP * HDIM);
    k_norm<<<g1((size_t)NFL * HDIM), 256, 0, stream>>>(acc_pp2fl, cnt_pp2fl, m_pp2fl, NFL * HDIM);
    k_norm<<<g1((size_t)NPP * HDIM), 256, 0, stream>>>(acc_fl2pp, cnt_fl2pp, m_fl2pp, NPP * HDIM);

    // GRU updates (h_pp updated twice, sequentially, as in the reference)
    k_gru<<<gip, gru_tpb, 0, stream>>>(m_pp2ip, h_ip, h_ip16,
        wih_ip16, whh_ip16, bih_ip, bhh_ip, tip);
    k_gru<<<gfl, gru_tpb, 0, stream>>>(m_pp2fl, h_fl, h_fl16,
        wih_fl16, whh_fl16, bih_fl, bhh_fl, tfl);
    k_gru<<<gpp, gru_tpb, 0, stream>>>(m_fl2pp, h_pp, h_pp16,
        wih_ip16, whh_ip16, bih_ip, bhh_ip, tpp);
    k_gru<<<gpp, gru_tpb, 0, stream>>>(m_ip2pp, h_pp, h_pp16,
        wih_ip16, whh_ip16, bih_ip, bhh_ip, tpp);
  }

  k_readout<<<dim3((tfl + 3) / 4), dim3(128), 0, stream>>>(h_fl16, W1h, b1, W2h, b2,
                                                           W3, b3, out, tfl);
  (void)in_sizes; (void)n_in; (void)out_size; (void)ws_size;
}
